// HyperbolicLMHead_25615184953594
// MI455X (gfx1250) — compile-verified, matrix operand order checked by reading, not run on hardware
//
#include <hip/hip_runtime.h>
#include <stdint.h>

#define DDIM 256
#define TM 128
#define TN 128
#define KC 64
#define NCHUNK (DDIM / KC)
#define MAX_NORM 0.99999f   // 1.0 - 1e-5
#define EPSV 1e-7f

typedef unsigned short u16;
typedef __attribute__((ext_vector_type(4)))  unsigned int u32x4;
typedef __attribute__((ext_vector_type(8)))  unsigned int u32x8;
typedef __attribute__((ext_vector_type(4)))  float        f32x4;
typedef __attribute__((ext_vector_type(16))) __bf16       v16bf;
typedef __attribute__((ext_vector_type(8)))  float        v8f;

union Frag { u32x4 q[2]; v16bf v; };

__device__ __forceinline__ u16 f2bf(float x) {
  unsigned int u = __builtin_bit_cast(unsigned int, x);
  u += 0x7FFFu + ((u >> 16) & 1u);          // round-to-nearest-even
  return (u16)(u >> 16);
}
__device__ __forceinline__ float bf2f(u16 s) {
  unsigned int u = ((unsigned int)s) << 16;
  return __builtin_bit_cast(float, u);
}
__device__ __forceinline__ unsigned int pack2(u16 lo, u16 hi) {
  return (unsigned int)lo | ((unsigned int)hi << 16);
}

// ---------------------------------------------------------------------------
// TDM issue: one tensor_load_to_lds per 128x64 bf16 tile.
// D# group0: count=1 | lds_addr | global_addr[56:0] | type=2.
// D# group1: data_size=2B, tensor_dim0=256, tensor_dim1=128,
//            tile_dim0=64, tile_dim1=128, tensor_dim0_stride=256.
// Groups 2/3 NULL (2D tensor). Tracked by TENSORcnt.
// ---------------------------------------------------------------------------
__device__ __forceinline__ void tdm_issue(unsigned int lds_off, const void* gp,
                                          u32x8 g1) {
  unsigned long long ga = (unsigned long long)(uintptr_t)gp;
  u32x4 g0;
  g0.x = 1u;                                             // count=1
  g0.y = lds_off;                                        // lds_addr
  g0.z = (unsigned int)ga;                               // global_addr[31:0]
  g0.w = (unsigned int)((ga >> 32) & 0x01FFFFFFu)        // global_addr[56:32]
         | (2u << 30);                                   // type=2 (image)
  asm volatile("tensor_load_to_lds %0, %1" : : "s"(g0), "s"(g1) : "memory");
}

// ---------------------------------------------------------------------------
// Kernel 1: project rows onto Poincare ball, emit bf16 hi/lo split + sq-norms.
// One wave32 per row of 256 floats (8 floats per lane).
// ---------------------------------------------------------------------------
__global__ void __launch_bounds__(256)
hyp_prep_kernel(const float* __restrict__ H, const float* __restrict__ Vemb,
                u16* __restrict__ hhi, u16* __restrict__ hlo, float* __restrict__ h2,
                u16* __restrict__ vhi, u16* __restrict__ vlo, float* __restrict__ v2,
                int NH, int NHp, int NV, int NVp) {
  int gw   = (int)((blockIdx.x * blockDim.x + threadIdx.x) >> 5);
  int lane = (int)(threadIdx.x & 31);
  if (gw >= NHp + NVp) return;

  const float* src; u16* dhi; u16* dlo; float* dn; bool valid;
  if (gw < NHp) {
    valid = gw < NH;
    src = H    + (size_t)gw * DDIM;
    dhi = hhi  + (size_t)gw * DDIM;
    dlo = hlo  + (size_t)gw * DDIM;
    dn  = h2 + gw;
  } else {
    int vr = gw - NHp;
    valid = vr < NV;
    src = Vemb + (size_t)vr * DDIM;
    dhi = vhi  + (size_t)vr * DDIM;
    dlo = vlo  + (size_t)vr * DDIM;
    dn  = v2 + vr;
  }

  float x[8];
  if (valid) {
    const f32x4* p = (const f32x4*)src;
    f32x4 a = p[lane * 2 + 0];
    f32x4 b = p[lane * 2 + 1];
    x[0] = a.x; x[1] = a.y; x[2] = a.z; x[3] = a.w;
    x[4] = b.x; x[5] = b.y; x[6] = b.z; x[7] = b.w;
  } else {
#pragma unroll
    for (int i = 0; i < 8; ++i) x[i] = 0.0f;
  }

  float s = 0.0f;
#pragma unroll
  for (int i = 0; i < 8; ++i) s += x[i] * x[i];
#pragma unroll
  for (int m = 16; m > 0; m >>= 1) s += __shfl_xor(s, m, 32);

  float norm  = __fsqrt_rn(s);
  float scale = (norm > MAX_NORM) ? (MAX_NORM / fmaxf(norm, EPSV)) : 1.0f;

  u16 hi[8]; u16 lo[8];
#pragma unroll
  for (int i = 0; i < 8; ++i) {
    float xv = x[i] * scale;
    hi[i] = f2bf(xv);
    lo[i] = f2bf(xv - bf2f(hi[i]));
  }
  u32x4 qh, ql;
  qh.x = pack2(hi[0], hi[1]); qh.y = pack2(hi[2], hi[3]);
  qh.z = pack2(hi[4], hi[5]); qh.w = pack2(hi[6], hi[7]);
  ql.x = pack2(lo[0], lo[1]); ql.y = pack2(lo[2], lo[3]);
  ql.z = pack2(lo[4], lo[5]); ql.w = pack2(lo[6], lo[7]);
  ((u32x4*)dhi)[lane] = qh;
  ((u32x4*)dlo)[lane] = ql;
  if (lane == 0) *dn = s * scale * scale;   // ||proj(x)||^2
}

// ---------------------------------------------------------------------------
// Kernel 2: 128x128 bf16x3 WMMA GEMM with TDM double-buffered LDS staging,
// hyperbolic-distance epilogue with non-temporal output stores.
// 8 waves/block; wave w owns rows [w*16, w*16+16) x 128 cols (8 acc tiles).
// ---------------------------------------------------------------------------
__global__ void __launch_bounds__(256)
hyp_gemm_kernel(const u16* __restrict__ hhi, const u16* __restrict__ hlo,
                const float* __restrict__ h2,
                const u16* __restrict__ vhi, const u16* __restrict__ vlo,
                const float* __restrict__ v2,
                float* __restrict__ out, int NH, int NV) {
  __shared__ u16 sAhi[2][TM * KC];
  __shared__ u16 sAlo[2][TM * KC];
  __shared__ u16 sBhi[2][TN * KC];
  __shared__ u16 sBlo[2][TN * KC];

  const int tid  = (int)threadIdx.x;
  const int wave = tid >> 5;
  const int lane = tid & 31;
  const int rowBase = (int)blockIdx.y * TM;
  const int colBase = (int)blockIdx.x * TN;

  // Shared D# group1 for all tile loads: 2-byte elems, 64x128 tile of a
  // row-major [rows x 256] matrix.
  u32x8 g1;
  g1[0] = 0x00010000u;        // workgroup_mask=0, data_size=1 (2 bytes)
  g1[1] = (unsigned)DDIM << 16;   // tensor_dim0 = 256 (low 16 in bits 63:48)
  g1[2] = 128u << 16;         // tensor_dim0 hi = 0; tensor_dim1 = 128
  g1[3] = (unsigned)KC << 16; // tensor_dim1 hi = 0; tile_dim0 = 64
  g1[4] = (unsigned)TM;       // tile_dim1 = 128, tile_dim2 = 0
  g1[5] = (unsigned)DDIM;     // tensor_dim0_stride = 256 (elems)
  g1[6] = 0u;
  g1[7] = 0u;

  auto issue = [&](int c, int buf) {
    int k0 = c * KC;
    tdm_issue((unsigned int)(uintptr_t)&sAhi[buf][0],
              hhi + (size_t)rowBase * DDIM + k0, g1);
    tdm_issue((unsigned int)(uintptr_t)&sAlo[buf][0],
              hlo + (size_t)rowBase * DDIM + k0, g1);
    tdm_issue((unsigned int)(uintptr_t)&sBhi[buf][0],
              vhi + (size_t)colBase * DDIM + k0, g1);
    tdm_issue((unsigned int)(uintptr_t)&sBlo[buf][0],
              vlo + (size_t)colBase * DDIM + k0, g1);
  };

  v8f acc[8] = {};
  const int m16 = lane & 15;
  const int kh  = lane >> 4;

  if (wave == 0) issue(0, 0);

  for (int c = 0; c < NCHUNK; ++c) {
    const int buf = c & 1;
    if (wave == 0) __builtin_amdgcn_s_wait_tensorcnt(0);  // chunk c landed
    __syncthreads();           // publish buf; all waves done with buf^1
    if (wave == 0 && (c + 1) < NCHUNK) issue(c + 1, buf ^ 1);  // prefetch

    const u16* pAhi = &sAhi[buf][0];
    const u16* pAlo = &sAlo[buf][0];
    const u16* pBhi = &sBhi[buf][0];
    const u16* pBlo = &sBlo[buf][0];

#pragma unroll
    for (int ks = 0; ks < KC; ks += 32) {
      // A fragment (16x32 bf16): lanes 0-15 hold K[0..7]+K[16..23] of row m,
      // lanes 16-31 hold K[8..15]+K[24..31] (ISA 7.12.2 layout).
      int aoff = (wave * 16 + m16) * KC + ks + kh * 8;
      Frag ahi, alo;
      ahi.q[0] = *(const u32x4*)(pAhi + aoff);
      ahi.q[1] = *(const u32x4*)(pAhi + aoff + 16);
      alo.q[0] = *(const u32x4*)(pAlo + aoff);
      alo.q[1] = *(const u32x4*)(pAlo + aoff + 16);
#pragma unroll
      for (int j = 0; j < 8; ++j) {
        int boff = (j * 16 + m16) * KC + ks + kh * 8;
        Frag bhi, blo;
        bhi.q[0] = *(const u32x4*)(pBhi + boff);
        bhi.q[1] = *(const u32x4*)(pBhi + boff + 16);
        blo.q[0] = *(const u32x4*)(pBlo + boff);
        blo.q[1] = *(const u32x4*)(pBlo + boff + 16);
        // bf16x3 split: hi*hi + hi*lo + lo*hi ~= fp32 dot product
        acc[j] = __builtin_amdgcn_wmma_f32_16x16x32_bf16(
            false, ahi.v, false, bhi.v, (short)0, acc[j], false, false);
        acc[j] = __builtin_amdgcn_wmma_f32_16x16x32_bf16(
            false, ahi.v, false, blo.v, (short)0, acc[j], false, false);
        acc[j] = __builtin_amdgcn_wmma_f32_16x16x32_bf16(
            false, alo.v, false, bhi.v, (short)0, acc[j], false, false);
      }
    }
  }

  // Epilogue. C/D layout: VGPR r, lane l -> M = r + 8*(l>>4), N = l&15.
  float h2v[8];
#pragma unroll
  for (int r = 0; r < 8; ++r) {
    int row = rowBase + wave * 16 + kh * 8 + r;
    h2v[r] = (row < NH) ? h2[row] : 0.0f;
  }
#pragma unroll
  for (int j = 0; j < 8; ++j) {
    int col = colBase + j * 16 + m16;
    if (col >= NV) continue;
    float v2c    = v2[col];
    float one_mv = 1.0f - v2c;
#pragma unroll
    for (int r = 0; r < 8; ++r) {
      int row = rowBase + wave * 16 + kh * 8 + r;
      if (row >= NH) continue;
      float dot   = acc[j][r];
      float sq    = fmaxf(h2v[r] + v2c - 2.0f * dot, 0.0f);
      float denom = (1.0f - h2v[r]) * one_mv;
      float arg   = 1.0f + 2.0f * sq / fmaxf(denom, EPSV);
      arg         = fmaxf(arg, 1.0f + 1e-7f);
      float dist  = __logf(arg + __fsqrt_rn(arg * arg - 1.0f));  // arccosh
      __builtin_nontemporal_store(-dist, &out[(size_t)row * NV + col]);
    }
  }
}

// ---------------------------------------------------------------------------
extern "C" void kernel_launch(void* const* d_in, const int* in_sizes, int n_in,
                              void* d_out, int out_size, void* d_ws, size_t ws_size,
                              hipStream_t stream) {
  (void)n_in; (void)out_size; (void)ws_size;
  const float* H    = (const float*)d_in[0];   // [N, 256] fp32
  const float* Vemb = (const float*)d_in[1];   // [V, 256] fp32
  int NH  = in_sizes[0] / DDIM;                // 4096
  int NV  = in_sizes[1] / DDIM;                // 50257
  int NHp = ((NH + TM - 1) / TM) * TM;
  int NVp = ((NV + TN - 1) / TN) * TN;

  char* ws = (char*)d_ws;
  size_t o = 0;
  u16*   hhi = (u16*)(ws + o);   o += (size_t)NHp * DDIM * 2;
  u16*   hlo = (u16*)(ws + o);   o += (size_t)NHp * DDIM * 2;
  u16*   vhi = (u16*)(ws + o);   o += (size_t)NVp * DDIM * 2;
  u16*   vlo = (u16*)(ws + o);   o += (size_t)NVp * DDIM * 2;
  float* h2  = (float*)(ws + o); o += (size_t)NHp * 4;
  float* v2  = (float*)(ws + o); o += (size_t)NVp * 4;

  int rows = NHp + NVp;                        // one wave32 per row
  int pb   = (rows + 7) / 8;                   // 8 waves per 256-thread block
  hyp_prep_kernel<<<pb, 256, 0, stream>>>(H, Vemb, hhi, hlo, h2, vhi, vlo, v2,
                                          NH, NHp, NV, NVp);

  dim3 grid(NVp / TN, NHp / TM);
  hyp_gemm_kernel<<<grid, 256, 0, stream>>>(hhi, hlo, h2, vhi, vlo, v2,
                                            (float*)d_out, NH, NV);
}
